// Recurrence_40338332844697
// MI455X (gfx1250) — matrix operand order repaired and checked by reading, not compiled, not run on hardware
//
#include <hip/hip_runtime.h>

#define TT 32
#define NN 64
#define CC 64
#define LL 64
#define HH 256
#define G3 768
#define HSTR 258   // padded hbuf row stride (floats): kills 16-way bank conflicts

typedef __attribute__((ext_vector_type(16))) __bf16 v16bf;
typedef __attribute__((ext_vector_type(8)))  float  v8f;

__device__ __forceinline__ float sigmf(float x){ return 1.f/(1.f+expf(-x)); }

// ---------------------------------------------------------------------------
// Pack a fp32 [768][256] weight W (g-major, gemm is  out[n,g] = sum_k in[n,k]*W[g,k])
// into WMMA-B bf16 tiles for v_wmma_f32_16x16x32_bf16:
//   B tile (kt,gt) is 32x16: lane holds column N = lane%16,
//   K = kt*32 + (e>=8 ? 16 : 0) + (lane>=16 ? 8 : 0) + (e&7)   (e = element 0..15, packed 2/VGPR).
// pack[gt*4096 + kt*512 + lane*16 + e] = W[g*256 + k]
// ---------------------------------------------------------------------------
__global__ void pack_b_bf16(const float* __restrict__ w, __bf16* __restrict__ dst){
  int idx  = blockIdx.x*256 + threadIdx.x;          // 196608 total
  int e    = idx & 15;
  int lane = (idx >> 4) & 31;
  int kt   = (idx >> 9) & 7;
  int gt   =  idx >> 12;                            // 0..47
  int g = gt*16 + (lane & 15);
  int k = kt*32 + ((e >> 3) << 4) + ((lane >> 4) << 3) + (e & 7);
  dst[idx] = (__bf16)w[g*256 + k];
}

__global__ void transpose_f32(const float* __restrict__ src, float* __restrict__ dst,
                              int rows, int cols){
  int idx = blockIdx.x*256 + threadIdx.x;
  if (idx < rows*cols){
    int r = idx / cols, c = idx - r*cols;
    dst[c*rows + r] = src[idx];
  }
}

// ---------------------------------------------------------------------------
// Phase 1: x[t,n,:] = embed[lines[n,t]];  xi_d[t,n,g] = sum_h x*wi_d[g,h] + bi_d[g]
// One wave per 16x16 output tile; 8 waves/block cover 8 g-tiles.
// grid = (6 gtGroups, 256 rowTiles, 2 dirs), block = 256.
// ---------------------------------------------------------------------------
__global__ void input_proj_kernel(const int* __restrict__ lines,
                                  const float* __restrict__ embed,
                                  const __bf16* __restrict__ wip_f,
                                  const __bf16* __restrict__ wip_b,
                                  const float* __restrict__ bi_f,
                                  const float* __restrict__ bi_b,
                                  float* __restrict__ xi_f,
                                  float* __restrict__ xi_b){
  int dir = blockIdx.z;
  const __bf16* wp = dir ? wip_b : wip_f;
  const float*  bi = dir ? bi_b  : bi_f;
  float*        xi = dir ? xi_b  : xi_f;
  int lane = threadIdx.x & 31, wave = threadIdx.x >> 5;
  int gt = blockIdx.x*8 + wave;                 // 0..47
  int rowTile = blockIdx.y;                     // 0..255 (L*N/16)
  int m = lane & 15;
  int r = rowTile*16 + m;                       // flat (t,n): r = t*64+n
  int t = r >> 6, n = r & 63;
  const float* arow = embed + (size_t)lines[n*LL + t]*HH;
  int hi8 = (lane >> 4) << 3;
  const __bf16* bp = wp + (size_t)gt*4096 + lane*16;
  v8f acc = {0.f,0.f,0.f,0.f,0.f,0.f,0.f,0.f};
  #pragma unroll
  for (int kt = 0; kt < 8; ++kt){
    v16bf a;
    #pragma unroll
    for (int v = 0; v < 8; ++v){
      int K0 = kt*32 + 2*(v&3) + ((v>>2)<<4) + hi8;   // ISA 16-bit A 16x32 swizzle
      float2 p = *(const float2*)(arow + K0);
      a[2*v] = (__bf16)p.x; a[2*v+1] = (__bf16)p.y;
    }
    v16bf b = *(const v16bf*)(bp + kt*512);
    acc = __builtin_amdgcn_wmma_f32_16x16x32_bf16(false, a, false, b,
                                                  (short)0, acc, false, false);
  }
  int g = gt*16 + (lane & 15);
  float bias = bi[g];
  int rbase = rowTile*16 + ((lane >> 4) << 3);  // D rows: v + (lane>=16)*8
  #pragma unroll
  for (int v = 0; v < 8; ++v) xi[(size_t)(rbase+v)*G3 + g] = acc[v] + bias;
}

// ---------------------------------------------------------------------------
// Phase 2: GRU recurrence bank.
// grid = (64 rolls, 2 dirs, 2 nTiles), block = 256 (8 waves).
// Block owns 32 batch rows (two 16-row A groups) so every B fragment loaded
// from L2 feeds TWO WMMAs -> halves B bandwidth per FLOP (the phase-2 limiter).
// A-fragments hoisted into registers once per step (2 x 64 VGPRs), reused
// across the wave's 6 g-column tiles: 96 WMMAs / wave-step.
// D tiles stored packed: ghbuf[grp*12288 + gt*256 + lane*8 + v] (2x ds_store_b128).
// hbuf rows padded to 258 floats to avoid 16-way LDS bank conflicts.
// K layout: K[n][roll][kk][h], kk = 2t (fwd) / 2*(63-t)+1 (bwd).
// ---------------------------------------------------------------------------
__global__ void gru_bank_kernel(const __bf16* __restrict__ whp_f,
                                const __bf16* __restrict__ whp_b,
                                const float* __restrict__ bh_f,
                                const float* __restrict__ bh_b,
                                const float* __restrict__ xi_f,
                                const float* __restrict__ xi_b,
                                __bf16* __restrict__ Kb){
  __shared__ float hbuf[32*HSTR];     // fp32 hidden state, 32 rows (padded stride)
  __shared__ float ghbuf[2*48*256];   // packed D tiles per row-group: [grp][gt][lane][v]
  int roll = blockIdx.x, dir = blockIdx.y, ntile = blockIdx.z;   // ntile: 0..1
  const __bf16* wp = dir ? whp_b : whp_f;
  const float*  bh = dir ? bh_b  : bh_f;
  const float*  xi = dir ? xi_b  : xi_f;
  int tid = threadIdx.x, wave = tid >> 5, lane = tid & 31;
  int m   = lane & 15;
  int hi8 = (lane >> 4) << 3;
  for (int i = tid; i < 32*HSTR; i += 256) hbuf[i] = 0.f;
  __syncthreads();
  for (int t = 0; t < LL; ++t){
    // source timestep of the rolled sequence (fwd) / reversed rolled sequence (bwd)
    int ts = (dir == 0) ? ((t - roll) & 63) : ((63 - t - roll) & 63);
    // ---- hoist A fragments (hidden state, both row groups) into registers ----
    v16bf afA[8], afB[8];
    #pragma unroll
    for (int kt = 0; kt < 8; ++kt){
      #pragma unroll
      for (int v = 0; v < 8; ++v){
        int K0 = kt*32 + 2*(v&3) + ((v>>2)<<4) + hi8;   // ISA 16-bit A 16x32 swizzle
        float2 pA = *(const float2*)&hbuf[ m      *HSTR + K0];
        float2 pB = *(const float2*)&hbuf[(16 + m)*HSTR + K0];
        afA[kt][2*v] = (__bf16)pA.x; afA[kt][2*v+1] = (__bf16)pA.y;
        afB[kt][2*v] = (__bf16)pB.x; afB[kt][2*v+1] = (__bf16)pB.y;
      }
    }
    // ---- gh = h @ wh.T : 6 column tiles per wave, B reused by both row groups ----
    #pragma unroll
    for (int gti = 0; gti < 6; ++gti){
      int gt = wave*6 + gti;
      const __bf16* bp = wp + (size_t)gt*4096 + lane*16;
      v8f accA = {0.f,0.f,0.f,0.f,0.f,0.f,0.f,0.f};
      v8f accB = {0.f,0.f,0.f,0.f,0.f,0.f,0.f,0.f};
      #pragma unroll
      for (int kt = 0; kt < 8; ++kt){
        v16bf b = *(const v16bf*)(bp + kt*512);
        accA = __builtin_amdgcn_wmma_f32_16x16x32_bf16(false, afA[kt], false, b,
                                                       (short)0, accA, false, false);
        accB = __builtin_amdgcn_wmma_f32_16x16x32_bf16(false, afB[kt], false, b,
                                                       (short)0, accB, false, false);
      }
      float4* gpA = (float4*)&ghbuf[gt*256 + lane*8];
      gpA[0] = make_float4(accA[0], accA[1], accA[2], accA[3]);
      gpA[1] = make_float4(accA[4], accA[5], accA[6], accA[7]);
      float4* gpB = (float4*)&ghbuf[12288 + gt*256 + lane*8];
      gpB[0] = make_float4(accB[0], accB[1], accB[2], accB[3]);
      gpB[1] = make_float4(accB[4], accB[5], accB[6], accB[7]);
    }
    __syncthreads();
    // ---- fp32 gates + K writeout (32 rows x 256 cols) ----
    int kk = (dir == 0) ? (2*t) : (2*(63 - t) + 1);
    for (int i = tid; i < 32*HH; i += 256){
      int r2 = i >> 8, c = i & 255;         // r2: 0..31, c: tid
      int grp = r2 >> 4, mr = r2 & 15;
      int ng = ntile*32 + r2;
      const float* gi = xi + ((size_t)ts*NN + ng)*G3;   // already includes bi
      // gh(mr, g): ghbuf[grp*12288 + (g>>4)*256 + ((g&15) + 16*(mr>=8))*8 + (mr&7)]
      int base = grp*12288;
      int gtr  = c >> 4;
      int lsel = ((c & 15) + ((mr >= 8) ? 16 : 0))*8 + (mr & 7);
      float hr = ghbuf[base + (     gtr)*256 + lsel] + bh[      c];
      float hz = ghbuf[base + (16 + gtr)*256 + lsel] + bh[256 + c];
      float hn = ghbuf[base + (32 + gtr)*256 + lsel] + bh[512 + c];
      float rg  = sigmf(gi[      c] + hr);
      float zg  = sigmf(gi[256 + c] + hz);
      float ngt = tanhf(gi[512 + c] + rg*hn);
      float hnew = (1.f - zg)*ngt + zg*hbuf[r2*HSTR + c];
      hbuf[r2*HSTR + c] = hnew;
      Kb[(((size_t)ng*LL + roll)*128 + kk)*HH + c] = (__bf16)hnew;
    }
    __syncthreads();
  }
}

// ---------------------------------------------------------------------------
// Phase 3: actor/critic scan. Independent per batch row -> 64 blocks x 256 thr.
// Exact fp32; weights pre-transposed to k-major for coalescing.
// out row: [a, p, v, h(256), a_probs(16), p_probs(64)] = 339 floats.
// ---------------------------------------------------------------------------
__global__ void scan_kernel(const float* __restrict__ condition,
                            const int* __restrict__ a_idx,
                            const int* __restrict__ p_idx,
                            const float* __restrict__ aemb,
                            const float* __restrict__ fw1t, const float* __restrict__ f_b1,
                            const float* __restrict__ fw2t, const float* __restrict__ f_b2,
                            const float* __restrict__ cwit, const float* __restrict__ cwht,
                            const float* __restrict__ cbi,  const float* __restrict__ cbh,
                            const float* __restrict__ critic_w, const float* __restrict__ critic_b,
                            const float* __restrict__ ptrt, const float* __restrict__ ptr_b,
                            const float* __restrict__ actt, const float* __restrict__ act_b,
                            const float* __restrict__ qryt, const float* __restrict__ qry_b,
                            const __bf16* __restrict__ Kb,
                            float* __restrict__ out){
  int n = blockIdx.x, tid = threadIdx.x;
  __shared__ float sh[256], sobs[320], sx1[256], sx2[256], sgi[768], sgh[768],
                   sq[256], sl[128], szz[256], spl[64], sal[16], sred[2];
  sh[tid] = 0.f;
  __syncthreads();
  for (int t = 0; t < TT; ++t){
    if (tid < 64) sobs[tid] = condition[((size_t)t*NN + n)*CC + tid];
    int ap = (t == 0) ? 0 : a_idx[(t-1)*NN + n];     // a_prev, zero index at t=0
    sobs[64 + tid] = aemb[(size_t)ap*HH + tid];
    __syncthreads();
    { float a2 = f_b1[tid];
      for (int k = 0; k < 320; ++k) a2 += sobs[k]*fw1t[k*256 + tid];
      sx1[tid] = fmaxf(a2, 0.f); }
    __syncthreads();
    { float a2 = f_b2[tid];
      for (int k = 0; k < 256; ++k) a2 += sx1[k]*fw2t[k*256 + tid];
      sx2[tid] = fmaxf(a2, 0.f); }
    __syncthreads();
    for (int g = tid; g < 768; g += 256){
      float ai = cbi[g], ah = cbh[g];
      for (int k = 0; k < 256; ++k){
        ai += sx2[k]*cwit[k*768 + g];
        ah += sh [k]*cwht[k*768 + g];
      }
      sgi[g] = ai; sgh[g] = ah;
    }
    __syncthreads();
    { float rg  = sigmf(sgi[tid]       + sgh[tid]);
      float zg  = sigmf(sgi[256 + tid] + sgh[256 + tid]);
      float nn2 = tanhf(sgi[512 + tid] + rg*sgh[512 + tid]);
      sh[tid] = (1.f - zg)*nn2 + zg*sh[tid]; }
    __syncthreads();
    if (tid < 64){
      float a2 = ptr_b[tid];
      for (int k = 0; k < 256; ++k) a2 += sh[k]*ptrt[k*64 + tid];
      spl[tid] = a2;
    }
    { float a2 = qry_b[tid];
      for (int k = 0; k < 256; ++k) a2 += sh[k]*qryt[k*256 + tid];
      sq[tid] = a2; }
    __syncthreads();
    if (tid == 0){
      float mx = spl[0];
      for (int j = 1; j < 64; ++j) mx = fmaxf(mx, spl[j]);
      float s = 0.f;
      for (int j = 0; j < 64; ++j){ spl[j] = expf(spl[j]-mx); s += spl[j]; }
      float inv = 1.f/s;
      for (int j = 0; j < 64; ++j) spl[j] *= inv;
    }
    int p = p_idx[t*NN + n];
    const __bf16* kb = Kb + (((size_t)n*LL + p)*128)*HH;   // [128][256] bf16
    __syncthreads();
    if (tid < 128){
      float a2 = 0.f;
      for (int c = 0; c < 256; ++c) a2 += (float)kb[tid*HH + c]*sq[c];
      sl[tid] = a2;
    }
    __syncthreads();
    { float a2 = 0.f;
      for (int kk2 = 0; kk2 < 128; ++kk2) a2 += sl[kk2]*(float)kb[kk2*HH + tid];
      szz[tid] = a2; }
    __syncthreads();
    if (tid < 16){
      float a2 = act_b[tid];
      for (int c = 0; c < 256; ++c) a2 += szz[c]*actt[c*16 + tid];
      sal[tid] = a2;
    } else if (tid == 16){
      float v = critic_b[0];
      for (int c = 0; c < 256; ++c) v += sh[c]*critic_w[c];
      sred[1] = v;
    }
    __syncthreads();
    if (tid == 0){
      float mx = sal[0];
      for (int j = 1; j < 16; ++j) mx = fmaxf(mx, sal[j]);
      float s = 0.f;
      for (int j = 0; j < 16; ++j){ sal[j] = expf(sal[j]-mx); s += sal[j]; }
      float inv = 1.f/s;
      for (int j = 0; j < 16; ++j) sal[j] *= inv;
    }
    __syncthreads();
    float* o = out + ((size_t)t*NN + n)*339;
    if (tid == 0){ o[0] = (float)a_idx[t*NN + n]; o[1] = (float)p; o[2] = sred[1]; }
    o[3 + tid] = sh[tid];
    if (tid < 16) o[259 + tid] = sal[tid];
    if (tid < 64) o[275 + tid] = spl[tid];
    __syncthreads();
  }
}

extern "C" void kernel_launch(void* const* d_in, const int* in_sizes, int n_in,
                              void* d_out, int out_size, void* d_ws, size_t ws_size,
                              hipStream_t stream){
  (void)in_sizes; (void)n_in; (void)out_size; (void)ws_size;
  const float* condition = (const float*)d_in[0];
  const int*   lines     = (const int*)  d_in[1];
  const int*   a_idx     = (const int*)  d_in[2];
  const int*   p_idx     = (const int*)  d_in[3];
  const float* embed     = (const float*)d_in[4];
  const float* aemb      = (const float*)d_in[5];
  const float* gru_wi_f  = (const float*)d_in[6];
  const float* gru_wh_f  = (const float*)d_in[7];
  const float* gru_bi_f  = (const float*)d_in[8];
  const float* gru_bh_f  = (const float*)d_in[9];
  const float* gru_wi_b  = (const float*)d_in[10];
  const float* gru_wh_b  = (const float*)d_in[11];
  const float* gru_bi_b  = (const float*)d_in[12];
  const float* gru_bh_b  = (const float*)d_in[13];
  const float* f_w1      = (const float*)d_in[14];
  const float* f_b1      = (const float*)d_in[15];
  const float* f_w2      = (const float*)d_in[16];
  const float* f_b2      = (const float*)d_in[17];
  const float* cell_wi   = (const float*)d_in[18];
  const float* cell_wh   = (const float*)d_in[19];
  const float* cell_bi   = (const float*)d_in[20];
  const float* cell_bh   = (const float*)d_in[21];
  const float* critic_w  = (const float*)d_in[22];
  const float* critic_b  = (const float*)d_in[23];
  const float* pointer_w = (const float*)d_in[24];
  const float* pointer_b = (const float*)d_in[25];
  const float* actor_w   = (const float*)d_in[26];
  const float* actor_b   = (const float*)d_in[27];
  const float* query_w   = (const float*)d_in[28];
  const float* query_b   = (const float*)d_in[29];
  float* out = (float*)d_out;

  char* wsp = (char*)d_ws;
  size_t off = 0;
  auto alloc = [&](size_t bytes)->void*{
    void* p = wsp + off; off += (bytes + 255) & ~(size_t)255; return p;
  };
  __bf16* Kb    = (__bf16*)alloc((size_t)64*64*128*256*2);  // 256 MB K bank (bf16)
  float*  xi_f  = (float*) alloc((size_t)64*64*768*4);      // 12 MB
  float*  xi_b  = (float*) alloc((size_t)64*64*768*4);
  __bf16* wip_f = (__bf16*)alloc(768*256*2);
  __bf16* wip_b = (__bf16*)alloc(768*256*2);
  __bf16* whp_f = (__bf16*)alloc(768*256*2);
  __bf16* whp_b = (__bf16*)alloc(768*256*2);
  float*  fw1t  = (float*) alloc(320*256*4);
  float*  fw2t  = (float*) alloc(256*256*4);
  float*  cwit  = (float*) alloc(768*256*4);
  float*  cwht  = (float*) alloc(768*256*4);
  float*  qryt  = (float*) alloc(256*256*4);
  float*  ptrt  = (float*) alloc(64*256*4);
  float*  actt  = (float*) alloc(16*256*4);

  // weight prep
  pack_b_bf16<<<768, 256, 0, stream>>>(gru_wi_f, wip_f);
  pack_b_bf16<<<768, 256, 0, stream>>>(gru_wi_b, wip_b);
  pack_b_bf16<<<768, 256, 0, stream>>>(gru_wh_f, whp_f);
  pack_b_bf16<<<768, 256, 0, stream>>>(gru_wh_b, whp_b);
  transpose_f32<<<320, 256, 0, stream>>>(f_w1,      fw1t, 256, 320);
  transpose_f32<<<256, 256, 0, stream>>>(f_w2,      fw2t, 256, 256);
  transpose_f32<<<768, 256, 0, stream>>>(cell_wi,   cwit, 768, 256);
  transpose_f32<<<768, 256, 0, stream>>>(cell_wh,   cwht, 768, 256);
  transpose_f32<<<256, 256, 0, stream>>>(query_w,   qryt, 256, 256);
  transpose_f32<<< 64, 256, 0, stream>>>(pointer_w, ptrt,  64, 256);
  transpose_f32<<< 16, 256, 0, stream>>>(actor_w,   actt,  16, 256);

  // phase 1: shared input projections (roll-invariant)
  input_proj_kernel<<<dim3(6, 256, 2), 256, 0, stream>>>(
      lines, embed, wip_f, wip_b, gru_bi_f, gru_bi_b, xi_f, xi_b);

  // phase 2: 256-way parallel GRU recurrence bank (WMMA, 2x B reuse)
  gru_bank_kernel<<<dim3(64, 2, 2), 256, 0, stream>>>(
      whp_f, whp_b, gru_bh_f, gru_bh_b, xi_f, xi_b, Kb);

  // phase 3: per-batch-row actor/critic scan (exact fp32)
  scan_kernel<<<64, 256, 0, stream>>>(
      condition, a_idx, p_idx, aemb, fw1t, f_b1, fw2t, f_b2,
      cwit, cwht, cell_bi, cell_bh, critic_w, critic_b,
      ptrt, pointer_b, actt, actor_b, qryt, query_b, Kb, out);
}
